// RNN_14637248545522
// MI455X (gfx1250) — matrix operand order repaired and checked by reading, not compiled, not run on hardware
//
#include <hip/hip_runtime.h>
#include <stdint.h>

// Seq2seq RNN (encoder scan + greedy decoder) for MI455X / gfx1250.
// Persistent block per 32-row batch tile; weights f16 in LDS; loop-invariant
// weight WMMA B-fragments hoisted into VGPRs; bias folded into the first WMMA's
// C operand; hidden state double-buffered in LDS; v_wmma_f32_16x16x32_f16 +
// hardware v_tanh_f32.

#define SEQ   128
#define EDIM  64
#define HDIM  128
#define VDIM  128
#define TDEC  20
#define START_TOK 3
#define BM    32          // batch rows per block
#define NTHREADS 256      // 8 waves: 2 row groups x 4 col groups

typedef __attribute__((ext_vector_type(16))) _Float16 v16h;
typedef __attribute__((ext_vector_type(8)))  float    v8f;

union Frag16 { v16h h; uint32_t u[8]; };

__device__ __forceinline__ float fast_tanh(float v) {
#if __has_builtin(__builtin_amdgcn_tanhf)
  return __builtin_amdgcn_tanhf(v);
#else
  return tanhf(v);
#endif
}

// A-matrix fragment (16x32 f16, contiguous k order, same as load_bfrag):
// lanes 0-15: row M=laneLo, K = 0..15 in VGPR0-7
// lanes 16-31: same rows, K = 16..31. rowp = start of this lane's row.
__device__ __forceinline__ v16h load_afrag(const _Float16* rowp, int kc, int laneHi) {
  const uint32_t* p = (const uint32_t*)rowp;
  const int b0 = kc * 16 + laneHi * 8;   // in u32 (= 2 halfs) units
  Frag16 f;
#pragma unroll
  for (int i = 0; i < 4; ++i) f.u[i] = p[b0 + i];
#pragma unroll
  for (int i = 0; i < 4; ++i) f.u[4 + i] = p[b0 + 4 + i];
  return f.h;
}

// B-matrix fragment (32x16 f16): lane's column = col, lanes 0-15 hold K=kc*32..+15,
// lanes 16-31 hold K=kc*32+16..+31 -> contiguous 16 halfs of w[col][*] (w row-major [N][K]).
__device__ __forceinline__ v16h load_bfrag(const _Float16* w, int ldk, int col, int kc, int laneHi) {
  const uint32_t* p = (const uint32_t*)(w + (size_t)col * ldk + kc * 32 + laneHi * 16);
  Frag16 f;
#pragma unroll
  for (int i = 0; i < 8; ++i) f.u[i] = p[i];
  return f.h;
}

__device__ __forceinline__ v8f wmma_f16(v16h a, v16h b, v8f c) {
  return __builtin_amdgcn_wmma_f32_16x16x32_f16(false, a, false, b, (short)0, c, false, false);
}

// LDS layout (bytes), total 214656 (< 320KB/WGP)
#define OFF_X      0          // BM*SEQ int       = 16384
#define OFF_EMBE   16384      // 128*64 f16       = 16384
#define OFF_WIHE   32768      // 128*64 f16       = 16384
#define OFF_WHHE   49152      // 128*128 f16      = 32768
#define OFF_EMBD   81920
#define OFF_WIHD   98304
#define OFF_WHHD   114688
#define OFF_FCW    147456
#define OFF_HBUF   180224     // 2*BM*HDIM f16    = 16384
#define OFF_BIASE  196608     // 128 f32
#define OFF_BIASD  197120
#define OFF_FCB    197632
#define OFF_LG     198144     // BM*VDIM f32      = 16384
#define OFF_TOK    214528     // BM int
#define SMEM_BYTES 214656

__global__ __launch_bounds__(NTHREADS)
void rnn_s2s_kernel(const int* __restrict__ x,
                    const float* __restrict__ encEmb, const float* __restrict__ encWih,
                    const float* __restrict__ encWhh, const float* __restrict__ encBih,
                    const float* __restrict__ encBhh,
                    const float* __restrict__ decEmb, const float* __restrict__ decWih,
                    const float* __restrict__ decWhh, const float* __restrict__ decBih,
                    const float* __restrict__ decBhh,
                    const float* __restrict__ fcW, const float* __restrict__ fcB,
                    float* __restrict__ out) {
  extern __shared__ char smem[];
  int*      x_lds = (int*)(smem + OFF_X);
  _Float16* embE  = (_Float16*)(smem + OFF_EMBE);
  _Float16* wihE  = (_Float16*)(smem + OFF_WIHE);
  _Float16* whhE  = (_Float16*)(smem + OFF_WHHE);
  _Float16* embD  = (_Float16*)(smem + OFF_EMBD);
  _Float16* wihD  = (_Float16*)(smem + OFF_WIHD);
  _Float16* whhD  = (_Float16*)(smem + OFF_WHHD);
  _Float16* fcw   = (_Float16*)(smem + OFF_FCW);
  _Float16* hbuf  = (_Float16*)(smem + OFF_HBUF);
  float*    biasE = (float*)(smem + OFF_BIASE);
  float*    biasD = (float*)(smem + OFF_BIASD);
  float*    fcb   = (float*)(smem + OFF_FCB);
  float*    lg    = (float*)(smem + OFF_LG);
  int*      tok   = (int*)(smem + OFF_TOK);

  const int tid  = threadIdx.x;
  const int row0 = blockIdx.x * BM;

  // ---- stage weights/embeddings (f32 -> f16), biases, tokens; zero h0 ----
  for (int i = tid; i < VDIM * EDIM; i += NTHREADS) {
    embE[i] = (_Float16)encEmb[i];
    wihE[i] = (_Float16)encWih[i];
    embD[i] = (_Float16)decEmb[i];
    wihD[i] = (_Float16)decWih[i];
  }
  for (int i = tid; i < HDIM * HDIM; i += NTHREADS) {
    whhE[i] = (_Float16)encWhh[i];
    whhD[i] = (_Float16)decWhh[i];
    fcw[i]  = (_Float16)fcW[i];
  }
  if (tid < HDIM) {
    biasE[tid] = encBih[tid] + encBhh[tid];
    biasD[tid] = decBih[tid] + decBhh[tid];
    fcb[tid]   = fcB[tid];
  }
  for (int i = tid; i < BM * SEQ; i += NTHREADS) x_lds[i] = x[(size_t)row0 * SEQ + i];
  for (int i = tid; i < BM * HDIM; i += NTHREADS) hbuf[i] = (_Float16)0.0f;
  if (tid < BM) tok[tid] = START_TOK;
  __syncthreads();

  const int wave    = tid >> 5;
  const int lane    = tid & 31;
  const int laneHi  = lane >> 4;
  const int laneLo  = lane & 15;
  const int mr      = (wave >> 2) * 16;   // row tile base within block (0 or 16)
  const int colBase = (wave & 3) * 32;    // two 16-col tiles per wave
  int p = 0;                              // h double-buffer parity

  // hoisted bias splats (v8f in registers; become the C operand of the first WMMA)
  v8f cE[2], cD[2], cF[2];
#pragma unroll
  for (int nt = 0; nt < 2; ++nt) {
    const float bE = biasE[colBase + nt * 16 + laneLo];
    const float bD = biasD[colBase + nt * 16 + laneLo];
    const float bF = fcb[colBase + nt * 16 + laneLo];
#pragma unroll
    for (int j = 0; j < 8; ++j) { cE[nt][j] = bE; cD[nt][j] = bD; cF[nt][j] = bF; }
  }

  // ---- hoist encoder weight B-fragments into registers (loop-invariant) ----
  v16h bWihE[2][2];   // [kc][nt]
  v16h bWhhE[4][2];
#pragma unroll
  for (int kc = 0; kc < 2; ++kc)
#pragma unroll
    for (int nt = 0; nt < 2; ++nt)
      bWihE[kc][nt] = load_bfrag(wihE, EDIM, colBase + nt * 16 + laneLo, kc, laneHi);
#pragma unroll
  for (int kc = 0; kc < 4; ++kc)
#pragma unroll
    for (int nt = 0; nt < 2; ++nt)
      bWhhE[kc][nt] = load_bfrag(whhE, HDIM, colBase + nt * 16 + laneLo, kc, laneHi);

  // ---------------- encoder scan ----------------
  for (int t = 0; t < SEQ; ++t) {
    v8f acc[2];
    // embedding contribution: A = enc_emb[x[row,t]]  (16 x 64); bias folded into kc=0
    const int tk = x_lds[(mr + laneLo) * SEQ + t];
    const _Float16* erow = embE + tk * EDIM;
    {
      const v16h a0 = load_afrag(erow, 0, laneHi);
#pragma unroll
      for (int nt = 0; nt < 2; ++nt) acc[nt] = wmma_f16(a0, bWihE[0][nt], cE[nt]);
      const v16h a1 = load_afrag(erow, 1, laneHi);
#pragma unroll
      for (int nt = 0; nt < 2; ++nt) acc[nt] = wmma_f16(a1, bWihE[1][nt], acc[nt]);
    }
    // recurrent contribution: A = h (16 x 128)
    const _Float16* hrow = hbuf + p * (BM * HDIM) + (mr + laneLo) * HDIM;
#pragma unroll
    for (int kc = 0; kc < 4; ++kc) {
      const v16h a = load_afrag(hrow, kc, laneHi);
#pragma unroll
      for (int nt = 0; nt < 2; ++nt) acc[nt] = wmma_f16(a, bWhhE[kc][nt], acc[nt]);
    }
    // tanh + store new h (f16) into other buffer
    _Float16* hdst = hbuf + (p ^ 1) * (BM * HDIM);
#pragma unroll
    for (int nt = 0; nt < 2; ++nt) {
      const int coln = colBase + nt * 16 + laneLo;
#pragma unroll
      for (int j = 0; j < 8; ++j)
        hdst[(mr + laneHi * 8 + j) * HDIM + coln] = (_Float16)fast_tanh(acc[nt][j]);
    }
    __syncthreads();
    p ^= 1;
  }

  // ---- hoist decoder recurrent weight B-fragments (fc stays in LDS) ----
  v16h bWihD[2][2];
  v16h bWhhD[4][2];
#pragma unroll
  for (int kc = 0; kc < 2; ++kc)
#pragma unroll
    for (int nt = 0; nt < 2; ++nt)
      bWihD[kc][nt] = load_bfrag(wihD, EDIM, colBase + nt * 16 + laneLo, kc, laneHi);
#pragma unroll
  for (int kc = 0; kc < 4; ++kc)
#pragma unroll
    for (int nt = 0; nt < 2; ++nt)
      bWhhD[kc][nt] = load_bfrag(whhD, HDIM, colBase + nt * 16 + laneLo, kc, laneHi);

  // ---------------- greedy decoder ----------------
  for (int t = 0; t < TDEC; ++t) {
    // h_new = tanh(dec_emb[tok] @ Wih^T + h @ Whh^T + bias)
    v8f acc[2];
    const int tk = tok[mr + laneLo];
    const _Float16* erow = embD + tk * EDIM;
    {
      const v16h a0 = load_afrag(erow, 0, laneHi);
#pragma unroll
      for (int nt = 0; nt < 2; ++nt) acc[nt] = wmma_f16(a0, bWihD[0][nt], cD[nt]);
      const v16h a1 = load_afrag(erow, 1, laneHi);
#pragma unroll
      for (int nt = 0; nt < 2; ++nt) acc[nt] = wmma_f16(a1, bWihD[1][nt], acc[nt]);
    }
    const _Float16* hrow = hbuf + p * (BM * HDIM) + (mr + laneLo) * HDIM;
#pragma unroll
    for (int kc = 0; kc < 4; ++kc) {
      const v16h a = load_afrag(hrow, kc, laneHi);
#pragma unroll
      for (int nt = 0; nt < 2; ++nt) acc[nt] = wmma_f16(a, bWhhD[kc][nt], acc[nt]);
    }
    _Float16* hdst = hbuf + (p ^ 1) * (BM * HDIM);
#pragma unroll
    for (int nt = 0; nt < 2; ++nt) {
      const int coln = colBase + nt * 16 + laneLo;
#pragma unroll
      for (int j = 0; j < 8; ++j)
        hdst[(mr + laneHi * 8 + j) * HDIM + coln] = (_Float16)fast_tanh(acc[nt][j]);
    }
    __syncthreads();
    p ^= 1;

    // logits = h_new @ fcW^T + fc_b   (fc B-fragments streamed from LDS)
    v8f la[2];
    const _Float16* hr2 = hbuf + p * (BM * HDIM) + (mr + laneLo) * HDIM;
    {
      const v16h a0 = load_afrag(hr2, 0, laneHi);
#pragma unroll
      for (int nt = 0; nt < 2; ++nt)
        la[nt] = wmma_f16(a0, load_bfrag(fcw, HDIM, colBase + nt * 16 + laneLo, 0, laneHi), cF[nt]);
    }
#pragma unroll
    for (int kc = 1; kc < 4; ++kc) {
      const v16h a = load_afrag(hr2, kc, laneHi);
#pragma unroll
      for (int nt = 0; nt < 2; ++nt)
        la[nt] = wmma_f16(a, load_bfrag(fcw, HDIM, colBase + nt * 16 + laneLo, kc, laneHi), la[nt]);
    }
    // write logits: LDS (for argmax) + global out [B, TDEC, VDIM]
#pragma unroll
    for (int nt = 0; nt < 2; ++nt) {
      const int coln = colBase + nt * 16 + laneLo;
#pragma unroll
      for (int j = 0; j < 8; ++j) {
        const int rw = mr + laneHi * 8 + j;
        const float v = la[nt][j];
        lg[rw * VDIM + coln] = v;
        out[((size_t)(row0 + rw) * TDEC + t) * VDIM + coln] = v;
      }
    }
    __syncthreads();

    // greedy argmax per row (first-max tie-break, matches jnp.argmax)
    if (tid < BM) {
      const float* rowl = lg + tid * VDIM;
      float m = rowl[0];
      int mi = 0;
#pragma unroll 4
      for (int v2 = 1; v2 < VDIM; ++v2) {
        const float vv = rowl[v2];
        if (vv > m) { m = vv; mi = v2; }
      }
      tok[tid] = mi;
    }
    __syncthreads();
  }
}

extern "C" void kernel_launch(void* const* d_in, const int* in_sizes, int n_in,
                              void* d_out, int out_size, void* d_ws, size_t ws_size,
                              hipStream_t stream) {
  const int Btot = in_sizes[0] / SEQ;            // 4096
  dim3 grid(Btot / BM), block(NTHREADS);
  rnn_s2s_kernel<<<grid, block, SMEM_BYTES, stream>>>(
      (const int*)d_in[0],
      (const float*)d_in[1], (const float*)d_in[2], (const float*)d_in[3],
      (const float*)d_in[4], (const float*)d_in[5],
      (const float*)d_in[6], (const float*)d_in[7], (const float*)d_in[8],
      (const float*)d_in[9], (const float*)d_in[10],
      (const float*)d_in[11], (const float*)d_in[12],
      (float*)d_out);
}
